// LmLSTM_23012434772527
// MI455X (gfx1250) — matrix operand order repaired and checked by reading, not compiled, not run on hardware
//
#include <hip/hip_runtime.h>
#include <hip/hip_bf16.h>
#include <stdint.h>

// ---------------------------------------------------------------------------
// CDNA5 (gfx1250) LSTM LM.  wave32, WMMA bf16 16x16x32, fp32 accumulate.
// Round 3: hybrid per-timestep barrier -- hardware cluster barrier
// (s_barrier_signal/wait -3) when the 16 recurrence workgroups are dispatched
// as a cluster, atomic sense barrier otherwise. GEMMs keep the pipelined
// double-buffered operand loads + global_prefetch + NT logits stores.
// ---------------------------------------------------------------------------

typedef __bf16 bf16_t;
typedef __attribute__((ext_vector_type(16))) __bf16 v16bf;
typedef __attribute__((ext_vector_type(8)))  __bf16 v8bf;
typedef __attribute__((ext_vector_type(8)))  float  v8f;

#define V_  32000
#define E_  512
#define H_  1024
#define B_  16
#define T_  256
#define N4H 4096            // 4*H
#define M_  (B_ * T_)       // 4096 rows (time-major: m = t*16 + b)

// ---- WMMA operand loaders (contiguous k = 16*(lane>>4) + e in A and B) -------------------
// A 16x32 bf16 tile, row-major source (row = lane&15; lane>=16 takes K+16 half)
__device__ __forceinline__ v16bf load_a16(const bf16_t* __restrict__ A,
                                          int lda, int kbase) {
    const int lane = threadIdx.x & 31;
    const int row  = lane & 15;
    const int hi   = lane >> 4;
    const bf16_t* p = A + (size_t)row * lda + kbase + hi * 16;
    v8bf lo = *(const v8bf*)(p);        // K = k0 .. k0+7
    v8bf hv = *(const v8bf*)(p + 8);    // K = k0+8 .. k0+15
    return __builtin_shufflevector(lo, hv, 0,1,2,3,4,5,6,7,
                                          8,9,10,11,12,13,14,15);
}

// B 32x16 bf16 tile from lane-packed weights: Bp[((nt*KT+kt)*32+lane)*16 + e]
__device__ __forceinline__ v16bf load_b16(const bf16_t* __restrict__ Bp,
                                          int nt, int kt, int KT) {
    const int lane = threadIdx.x & 31;
    const bf16_t* p = Bp + ((((size_t)nt * KT + kt) * 32) + lane) * 16;
    return *(const v16bf*)p;   // 32B contiguous per lane
}

__device__ __forceinline__ v8f wmma_bf16(v16bf a, v16bf b, v8f c) {
    return __builtin_amdgcn_wmma_f32_16x16x32_bf16(
        /*neg_a=*/false, a, /*neg_b=*/false, b,
        /*c_mod=*/(short)0, c, /*reuse_a=*/false, /*reuse_b=*/false);
}

__device__ __forceinline__ float sigm(float x) {
    return 1.0f / (1.0f + __expf(-x));
}

// ---- cluster detection: IB_STS2.CLUSTER_ID[9:6] (0 = not in a cluster) ----
__device__ __forceinline__ bool in_cluster() {
    // s_getreg_b32 hwreg(HW_REG_IB_STS2, offset=6, size=4)
    // SIMM16 = id(28) | offset<<6 | (size-1)<<11
    return __builtin_amdgcn_s_getreg(28 | (6 << 6) | (3 << 11)) != 0;
}

// ---- grid-wide barrier across the 16 recurrence workgroups -----------------
// Cluster dispatch: ISA-recommended split barrier (one wave per WG signals,
// all waves wait).  Non-cluster dispatch: atomic sense barrier + s_sleep spin.
__device__ __forceinline__ void gridbar(unsigned* cnt, volatile unsigned* gen,
                                        unsigned& tgt, unsigned nb, bool clus) {
    __syncthreads();
    if (clus) {
        __threadfence();            // make h stores device-visible
        if (threadIdx.x == 0)
            asm volatile("s_barrier_signal -3" ::: "memory");
        asm volatile("s_barrier_wait -3" ::: "memory");
    } else if (threadIdx.x == 0) {
        ++tgt;
        __threadfence();
        if (atomicAdd(cnt, 1u) == nb - 1u) {
            atomicExch(cnt, 0u);
            __threadfence();
            atomicAdd((unsigned*)gen, 1u);
        } else {
            while (*gen < tgt) __builtin_amdgcn_s_sleep(2);
        }
        __threadfence();
    }
    __syncthreads();
}

// ---------------------------------------------------------------------------
// Repack fp32 [K,N] weight -> lane-packed bf16 for load_b16.
// dest idx d = ((nt*KT + kt)*32 + lane)*16 + e ; n = nt*16+(lane&15),
// k = kt*32 + (lane>>4)*16 + e  (contiguous k order, as load_a16).
// ---------------------------------------------------------------------------
__global__ void __launch_bounds__(256) repack_kernel(
    const float* __restrict__ W, bf16_t* __restrict__ Bp, int Ndim, int KT) {
    size_t d = (size_t)blockIdx.x * 256 + threadIdx.x;
    size_t total = (size_t)Ndim * KT * 32;   // = Ndim*K, the element count
    total *= 16;
    if (d >= total) return;
    int    e    = (int)(d & 15);
    int    lane = (int)((d >> 4) & 31);
    size_t rest = d >> 9;
    int    kt   = (int)(rest % (size_t)KT);
    int    nt   = (int)(rest / (size_t)KT);
    int    n    = nt * 16 + (lane & 15);
    int    k    = kt * 32 + (lane >> 4) * 16 + e;
    Bp[d] = (bf16_t)W[(size_t)k * Ndim + n];
}

// ---------------------------------------------------------------------------
// Embedding gather -> time-major bf16 x[T*B, E]; mask[T*B] as float.
// ---------------------------------------------------------------------------
__global__ void __launch_bounds__(256) embed_kernel(
    const int* __restrict__ tokens, const float* __restrict__ emb,
    bf16_t* __restrict__ x_tb, float* __restrict__ maskf) {
    size_t idx = (size_t)blockIdx.x * 256 + threadIdx.x;
    if (idx >= (size_t)M_ * E_) return;
    int e = (int)(idx % E_);
    int m = (int)(idx / E_);        // m = t*16 + b
    int t = m >> 4, b = m & 15;
    int tok = tokens[b * T_ + t];
    x_tb[(size_t)m * E_ + e] = (bf16_t)emb[(size_t)tok * E_ + e];
    if (e == 0) maskf[m] = (tok != 0) ? 1.0f : 0.0f;
}

// ---------------------------------------------------------------------------
// Pipelined WMMA strip: acc[4] += A[16xK] @ B[Kx64] for one wave.
// Double-buffers A and all 4 B tiles one K-step ahead so s_wait_loadcnt
// only waits on the older load group (memory/matrix overlap).
// ---------------------------------------------------------------------------
__device__ __forceinline__ void wmma_strip64(
    const bf16_t* __restrict__ Arow, int lda,
    const bf16_t* __restrict__ Bp, int n0tile, int KT, v8f acc[4]) {
    v16bf a_cur = load_a16(Arow, lda, 0);
    v16bf b_cur0 = load_b16(Bp, n0tile + 0, 0, KT);
    v16bf b_cur1 = load_b16(Bp, n0tile + 1, 0, KT);
    v16bf b_cur2 = load_b16(Bp, n0tile + 2, 0, KT);
    v16bf b_cur3 = load_b16(Bp, n0tile + 3, 0, KT);
#pragma unroll 2
    for (int kt = 0; kt < KT - 1; ++kt) {
        // issue next-iteration loads before consuming current tiles
        v16bf a_nxt  = load_a16(Arow, lda, (kt + 1) * 32);
        v16bf b_nxt0 = load_b16(Bp, n0tile + 0, kt + 1, KT);
        v16bf b_nxt1 = load_b16(Bp, n0tile + 1, kt + 1, KT);
        v16bf b_nxt2 = load_b16(Bp, n0tile + 2, kt + 1, KT);
        v16bf b_nxt3 = load_b16(Bp, n0tile + 3, kt + 1, KT);
        if (kt + 6 < KT) {  // prefetch the B weight stream ahead of the pipe
            const int lane = threadIdx.x & 31;
            __builtin_prefetch(
                Bp + ((((size_t)n0tile * KT + (kt + 6)) * 32) + lane) * 16,
                0, 3);
        }
        acc[0] = wmma_bf16(a_cur, b_cur0, acc[0]);
        acc[1] = wmma_bf16(a_cur, b_cur1, acc[1]);
        acc[2] = wmma_bf16(a_cur, b_cur2, acc[2]);
        acc[3] = wmma_bf16(a_cur, b_cur3, acc[3]);
        a_cur = a_nxt;
        b_cur0 = b_nxt0; b_cur1 = b_nxt1; b_cur2 = b_nxt2; b_cur3 = b_nxt3;
    }
    acc[0] = wmma_bf16(a_cur, b_cur0, acc[0]);
    acc[1] = wmma_bf16(a_cur, b_cur1, acc[1]);
    acc[2] = wmma_bf16(a_cur, b_cur2, acc[2]);
    acc[3] = wmma_bf16(a_cur, b_cur3, acc[3]);
}

// ---------------------------------------------------------------------------
// C[M,N] = A[M,K](bf16) @ Bp(packed bf16) + bias.  Block = 128M x 64N,
// 8 waves, each wave 16x64 (4 accumulators), pipelined K loop.
// ---------------------------------------------------------------------------
__global__ void __launch_bounds__(256) gemm_bias_kernel(
    const bf16_t* __restrict__ A, const bf16_t* __restrict__ Bp,
    const float* __restrict__ bias, float* __restrict__ C,
    int Ndim, int Kdim) {
    const int KT     = Kdim >> 5;
    const int wave   = threadIdx.x >> 5;
    const int lane   = threadIdx.x & 31;
    const int n0tile = blockIdx.x * 4;
    const int m0     = blockIdx.y * 128 + wave * 16;
    const bf16_t* Arow = A + (size_t)m0 * Kdim;
    v8f acc[4] = {};
    wmma_strip64(Arow, Kdim, Bp, n0tile, KT, acc);
    const int row = lane & 15, hi = lane >> 4;
#pragma unroll
    for (int j = 0; j < 4; ++j) {
        int n = (n0tile + j) * 16 + row;
        float bn = bias[n];
#pragma unroll
        for (int r = 0; r < 8; ++r) {
            int m = m0 + r + hi * 8;
            C[(size_t)m * Ndim + n] = acc[j][r] + bn;
        }
    }
}

// ---------------------------------------------------------------------------
// Logits GEMM with masked one-hot epilogue, writes [B,T,V] fp32 directly.
// Output is write-once (host-read) 524MB -> non-temporal stores so it does
// not evict the L2-resident weights/activations.
// ---------------------------------------------------------------------------
__global__ void __launch_bounds__(256) gemm_logits_kernel(
    const bf16_t* __restrict__ A, const bf16_t* __restrict__ Bp,
    const float* __restrict__ bout, const float* __restrict__ maskf,
    float* __restrict__ out, int Kdim) {
    const int KT     = Kdim >> 5;
    const int wave   = threadIdx.x >> 5;
    const int lane   = threadIdx.x & 31;
    const int n0tile = blockIdx.x * 4;
    const int m0     = blockIdx.y * 128 + wave * 16;
    const bf16_t* Arow = A + (size_t)m0 * Kdim;
    v8f acc[4] = {};
    wmma_strip64(Arow, Kdim, Bp, n0tile, KT, acc);
    const int row = lane & 15, hi = lane >> 4;
#pragma unroll
    for (int j = 0; j < 4; ++j) {
        int n = (n0tile + j) * 16 + row;
        float bn = bout[n];
#pragma unroll
        for (int r = 0; r < 8; ++r) {
            int m  = m0 + r + hi * 8;     // m = t*16 + b
            int t  = m >> 4, b = m & 15;
            float mt  = maskf[m];
            float val = acc[j][r] + bn;
            if (mt == 0.0f) val = (n == 0) ? 1.0f : 0.0f;   // onehot0 row
            __builtin_nontemporal_store(val, &out[((size_t)b * T_ + t) * V_ + n]);
        }
    }
}

// ---------------------------------------------------------------------------
// LSTM recurrence. 16 blocks x 256 threads. Block `blk` owns H-cols
// [blk*64, blk*64+64): per step computes z = h @ Wh for its 4x64 gate cols
// (16 WMMA tiles, K=1024), fuses gates/state update, ping-pongs h through
// global (hbuf[2][16][H]), c stays in registers. Per-step barrier: hardware
// cluster barrier when dispatched as a 16-WG cluster, atomic otherwise.
// ---------------------------------------------------------------------------
__global__ void __launch_bounds__(256) lstm_rec_kernel(
    const float* __restrict__ xz,      // [T,B,4H] fp32 (bias already added)
    const bf16_t* __restrict__ Whp,    // packed [K=1024 -> KT=32, N=4096]
    const float* __restrict__ maskf,   // [T*B]
    bf16_t* __restrict__ hbuf,         // [2][16][H]
    bf16_t* __restrict__ hseq,         // [T,B,H] output (post-mask h)
    unsigned* cnt, volatile unsigned* gen) {
    __shared__ float zs[16][260];      // 16 batch x 256 gate-cols (+pad)
    const int blk  = blockIdx.x;       // 0..15
    const int tid  = threadIdx.x;
    const int wave = tid >> 5, lane = tid & 31;
    const int g    = wave >> 1, half = wave & 1;    // gate, n-half
    const int row  = lane & 15, hi = lane >> 4;
    const bool clus = in_cluster();
    unsigned tgt = 0;

    // zero own slice of both h ping-pong buffers
    for (int i = tid; i < 2048; i += 256) {
        int buf = i >> 10, rem = i & 1023;
        int b = rem >> 6, j = rem & 63;
        hbuf[((size_t)buf * 16 + b) * H_ + blk * 64 + j] = (bf16_t)0.0f;
    }
    float creg[4] = {0.0f, 0.0f, 0.0f, 0.0f};
    gridbar(cnt, gen, tgt, 16, clus);

    const int ntbase = g * 64 + blk * 4 + half * 2;  // N4H/16 tile index

    for (int t = 0; t < T_; ++t) {
        const bf16_t* hcur = hbuf + (size_t)(t & 1) * 16 * H_;
        bf16_t*       hnxt = hbuf + (size_t)((t + 1) & 1) * 16 * H_;

        // ---- WMMA phase: two 16x16 z-tiles per wave, K=1024, pipelined ----
        v8f acc0 = {}, acc1 = {};
        v16bf a_cur  = load_a16(hcur, H_, 0);
        v16bf b_cur0 = load_b16(Whp, ntbase,     0, 32);
        v16bf b_cur1 = load_b16(Whp, ntbase + 1, 0, 32);
#pragma unroll 2
        for (int kt = 0; kt < 31; ++kt) {
            v16bf a_nxt  = load_a16(hcur, H_, (kt + 1) * 32);
            v16bf b_nxt0 = load_b16(Whp, ntbase,     kt + 1, 32);
            v16bf b_nxt1 = load_b16(Whp, ntbase + 1, kt + 1, 32);
            acc0 = wmma_bf16(a_cur, b_cur0, acc0);
            acc1 = wmma_bf16(a_cur, b_cur1, acc1);
            a_cur = a_nxt; b_cur0 = b_nxt0; b_cur1 = b_nxt1;
        }
        acc0 = wmma_bf16(a_cur, b_cur0, acc0);
        acc1 = wmma_bf16(a_cur, b_cur1, acc1);

#pragma unroll
        for (int r = 0; r < 8; ++r) {
            int m = r + hi * 8;                       // batch row
            zs[m][g * 64 + half * 32 + row]      = acc0[r];
            zs[m][g * 64 + half * 32 + 16 + row] = acc1[r];
        }
        __syncthreads();

        // ---- element phase: gates + state update (4 elems/thread) ----
        const float* xzt = xz + (size_t)t * 16 * N4H;
#pragma unroll
        for (int k = 0; k < 4; ++k) {
            int b  = (tid >> 6) + 4 * k;
            int j  = tid & 63;
            int bj = blk * 64 + j;
            const float* xr = xzt + (size_t)b * N4H;
            float iv = sigm (zs[b][0 * 64 + j] + xr[0 * H_ + bj]);
            float fv = sigm (zs[b][1 * 64 + j] + xr[1 * H_ + bj]);
            float gv = tanhf(zs[b][2 * 64 + j] + xr[2 * H_ + bj]);
            float ov = sigm (zs[b][3 * 64 + j] + xr[3 * H_ + bj]);
            float cn = fv * creg[k] + iv * gv;
            float hn = ov * tanhf(cn);
            float mt = maskf[t * 16 + b];
            float h2;
            if (mt != 0.0f) { creg[k] = cn; h2 = hn; }
            else            { h2 = (float)hcur[(size_t)b * H_ + bj]; }
            bf16_t hb = (bf16_t)h2;
            hnxt[(size_t)b * H_ + bj] = hb;
            hseq[((size_t)t * 16 + b) * H_ + bj] = hb;
        }
        __syncthreads();
        gridbar(cnt, gen, tgt, 16, clus);
    }
}

__global__ void zero_ctrl_kernel(unsigned* c) {
    if (threadIdx.x < 64) c[threadIdx.x] = 0u;
}

// ---------------------------------------------------------------------------
extern "C" void kernel_launch(void* const* d_in, const int* in_sizes, int n_in,
                              void* d_out, int out_size, void* d_ws,
                              size_t ws_size, hipStream_t stream) {
    (void)in_sizes; (void)n_in; (void)out_size; (void)ws_size;
    const int*   tokens = (const int*)  d_in[0];
    const float* emb    = (const float*)d_in[1];
    const float* Wx0    = (const float*)d_in[2];
    const float* Wh0    = (const float*)d_in[3];
    const float* b0     = (const float*)d_in[4];
    const float* Wx1    = (const float*)d_in[5];
    const float* Wh1    = (const float*)d_in[6];
    const float* b1     = (const float*)d_in[7];
    const float* Wout   = (const float*)d_in[8];
    const float* bout   = (const float*)d_in[9];
    float* out = (float*)d_out;

    uint8_t* p = (uint8_t*)d_ws;
    auto carve = [&](size_t bytes) -> uint8_t* {
        uint8_t* r = p; p += (bytes + 255) & ~(size_t)255; return r;
    };
    unsigned* ctrl  = (unsigned*)carve(256);
    bf16_t*   x_tb  = (bf16_t*) carve((size_t)M_ * E_ * 2);
    float*    maskf = (float*)  carve((size_t)M_ * 4);
    bf16_t*   Wx0p  = (bf16_t*) carve((size_t)E_ * N4H * 2);
    bf16_t*   Wh0p  = (bf16_t*) carve((size_t)H_ * N4H * 2);
    bf16_t*   Wx1p  = (bf16_t*) carve((size_t)H_ * N4H * 2);
    bf16_t*   Wh1p  = (bf16_t*) carve((size_t)H_ * N4H * 2);
    bf16_t*   Woutp = (bf16_t*) carve((size_t)H_ * V_ * 2);
    float*    xzbuf = (float*)  carve((size_t)M_ * N4H * 4);
    bf16_t*   hseq0 = (bf16_t*) carve((size_t)M_ * H_ * 2);
    bf16_t*   hseq1 = (bf16_t*) carve((size_t)M_ * H_ * 2);
    bf16_t*   hbuf  = (bf16_t*) carve((size_t)2 * 16 * H_ * 2);

    // weight repacks (bf16, WMMA lane layout)
    repack_kernel<<<(unsigned)(((size_t)E_ * N4H + 255) / 256), 256, 0, stream>>>(Wx0, Wx0p, N4H, E_ / 32);
    repack_kernel<<<(unsigned)(((size_t)H_ * N4H + 255) / 256), 256, 0, stream>>>(Wh0, Wh0p, N4H, H_ / 32);
    repack_kernel<<<(unsigned)(((size_t)H_ * N4H + 255) / 256), 256, 0, stream>>>(Wx1, Wx1p, N4H, H_ / 32);
    repack_kernel<<<(unsigned)(((size_t)H_ * N4H + 255) / 256), 256, 0, stream>>>(Wh1, Wh1p, N4H, H_ / 32);
    repack_kernel<<<(unsigned)(((size_t)H_ * V_  + 255) / 256), 256, 0, stream>>>(Wout, Woutp, V_, H_ / 32);

    // embedding + mask (time-major)
    embed_kernel<<<(unsigned)(((size_t)M_ * E_ + 255) / 256), 256, 0, stream>>>(tokens, emb, x_tb, maskf);

    // layer 0: xz = x @ Wx0 + b0 ; recurrence
    gemm_bias_kernel<<<dim3(N4H / 64, M_ / 128), 256, 0, stream>>>(x_tb, Wx0p, b0, xzbuf, N4H, E_);
    zero_ctrl_kernel<<<1, 64, 0, stream>>>(ctrl);
    lstm_rec_kernel<<<16, 256, 0, stream>>>(xzbuf, Wh0p, maskf, hbuf, hseq0, ctrl, ctrl + 8);

    // layer 1
    gemm_bias_kernel<<<dim3(N4H / 64, M_ / 128), 256, 0, stream>>>(hseq0, Wx1p, b1, xzbuf, N4H, H_);
    zero_ctrl_kernel<<<1, 64, 0, stream>>>(ctrl);
    lstm_rec_kernel<<<16, 256, 0, stream>>>(xzbuf, Wh1p, maskf, hbuf, hseq1, ctrl, ctrl + 8);

    // logits + mask epilogue -> [B,T,V]
    gemm_logits_kernel<<<dim3(V_ / 64, M_ / 128), 256, 0, stream>>>(hseq1, Woutp, bout, maskf, out, H_);
}
